// ARMonocularModel_76630806495420
// MI455X (gfx1250) — matrix operand order repaired and checked by reading, not compiled, not run on hardware
//
#include <hip/hip_runtime.h>
#include <math.h>

// ---------------------------------------------------------------------------
// Model constants (from reference)
// ---------------------------------------------------------------------------
#define BATCH   2
#define DMODEL  768
#define NHEAD   16
#define HDIM    48
#define NLAYER  3
#define N_IMG   1024
#define N_PAST  16
#define T_DEC   20
#define S0      (N_IMG + 1 + N_PAST)   // 1041
#define SMAX    (S0 + T_DEC)           // 1061
#define M0      (BATCH * S0)           // 2082
#define ATTN_SCALE 0.14433756729740643f // 1/sqrt(48)

typedef __attribute__((ext_vector_type(16))) __bf16 v16bf;
typedef __attribute__((ext_vector_type(8)))  float  v8f;

// GEMM / GEMV epilogue modes
#define GM_QKV   0   // scatter into Q buffer + K/V caches
#define GM_RESID 1   // fp32 residual add:  out[m*N+n] += acc + bias
#define GM_RELU  2   // relu; GEMM writes bf16, GEMV writes fp32

// CDNA5 async global->LDS copy (guarded: falls back to sync staging if the
// builtin is not available in this toolchain).
#if defined(__AMDGCN__) && __has_builtin(__builtin_amdgcn_global_load_async_to_lds_b128)
#define HAS_ASYNC_LDS 1
typedef int v4i_async __attribute__((vector_size(16)));
typedef __attribute__((address_space(1))) v4i_async* gas_v4i;
typedef __attribute__((address_space(3))) v4i_async* las_v4i;
#else
#define HAS_ASYNC_LDS 0
#endif

// ---------------------------------------------------------------------------
// fp32 -> bf16 bulk convert (weights)
// ---------------------------------------------------------------------------
__global__ __launch_bounds__(256) void k_f2bf(const float* __restrict__ src,
                                              __bf16* __restrict__ dst,
                                              long n) {
  long i = (long)blockIdx.x * blockDim.x + threadIdx.x;
  long stride = (long)gridDim.x * blockDim.x;
  for (; i < n; i += stride) dst[i] = (__bf16)src[i];
}

// ---------------------------------------------------------------------------
// Build the initial token sequence: [img tokens | intent token | past tokens]
// ---------------------------------------------------------------------------
__global__ __launch_bounds__(256) void k_embed(
    const float* __restrict__ feats,     // (B, 768, 1024)
    const float* __restrict__ past,      // (B, 16, 6)
    const int*   __restrict__ intent,    // (B,)
    const float* __restrict__ img_pos,   // (1, 1024, 768)
    const float* __restrict__ time_e,    // (36, 768)
    const float* __restrict__ intent_e,  // (3, 768)
    const float* __restrict__ past_W,    // (6, 768)
    const float* __restrict__ past_b,    // (768,)
    float* __restrict__ x) {
  const int s = blockIdx.x;
  const int b = blockIdx.y;
  float* xr = x + ((size_t)b * S0 + s) * DMODEL;
  for (int d = threadIdx.x; d < DMODEL; d += 256) {
    float v;
    if (s < N_IMG) {
      v = feats[((size_t)b * DMODEL + d) * N_IMG + s] + img_pos[(size_t)s * DMODEL + d];
    } else if (s == N_IMG) {
      int ii = intent[b] - 1;
      ii = ii < 0 ? 0 : (ii > 2 ? 2 : ii);
      v = intent_e[(size_t)ii * DMODEL + d];
    } else {
      int j = s - (N_IMG + 1);
      float a = past_b[d] + time_e[(size_t)j * DMODEL + d];
      #pragma unroll
      for (int c = 0; c < 6; ++c)
        a += past[((size_t)b * N_PAST + j) * 6 + c] * past_W[(size_t)c * DMODEL + d];
      v = a;
    }
    xr[d] = v;
  }
}

// ---------------------------------------------------------------------------
// LayerNorm over D=768; one block per token row.
// ---------------------------------------------------------------------------
template <typename OT>
__global__ __launch_bounds__(256) void k_ln(const float* __restrict__ x,
                                            const float* __restrict__ g,
                                            const float* __restrict__ beta,
                                            OT* __restrict__ out) {
  const int m = blockIdx.x;
  const int tid = threadIdx.x;
  const float* xr = x + (size_t)m * DMODEL;
  __shared__ float red[256];

  float s = 0.f;
  for (int d = tid; d < DMODEL; d += 256) s += xr[d];
  red[tid] = s;
  __syncthreads();
  for (int off = 128; off > 0; off >>= 1) {
    if (tid < off) red[tid] += red[tid + off];
    __syncthreads();
  }
  const float mean = red[0] * (1.f / DMODEL);
  __syncthreads();

  float v = 0.f;
  for (int d = tid; d < DMODEL; d += 256) {
    float t = xr[d] - mean;
    v += t * t;
  }
  red[tid] = v;
  __syncthreads();
  for (int off = 128; off > 0; off >>= 1) {
    if (tid < off) red[tid] += red[tid + off];
    __syncthreads();
  }
  const float rstd = rsqrtf(red[0] * (1.f / DMODEL) + 1e-5f);

  OT* orow = out + (size_t)m * DMODEL;
  for (int d = tid; d < DMODEL; d += 256)
    orow[d] = (OT)((xr[d] - mean) * rstd * g[d] + beta[d]);
}

// ---------------------------------------------------------------------------
// Tiled bf16 WMMA GEMM:  C[M,NN] = A[M,K] * B[K,NN] + bias, fused epilogue.
//   Templated on NN (= ldb) and MODE so the 16 strided B loads use immediate
//   offsets from one base pointer (no serialized address chain / xcnt waits)
//   and the epilogue branches fold away.
//   block = 128 threads (4 waves), macro tile = 128(M) x 64(N), K step = 32.
//   Wave w owns M strips {w, w+4}: 8 WMMAs per K-step, B fragments reused.
//   LDS layouts are swizzled so every WMMA fragment is contiguous 16B chunks
//   (2 x ds_load_b128 per fragment) per CDNA5 ISA 7.12.2 wave32 layouts.
//   A staging uses CDNA5 global_load_async_to_lds_b128; out-of-range rows are
//   CLAMPED (not branched) -- masked rows are discarded in the epilogue.
// ---------------------------------------------------------------------------
#define TM 128
#define TN 64
#define TK 32
#define A_STRIDE (TK + 8)   // 40 halfs = 80B, multiple of 16B
#define B_STRIDE 24         // halfs per fragment slot (48B, multiple of 16B)

template <int NN, int MODE>
__global__ __launch_bounds__(128) void k_gemm(
    const __bf16* __restrict__ A, int lda,
    const __bf16* __restrict__ Bw,
    const float* __restrict__ bias,
    int M, int K,
    float*  __restrict__ resid_out,   // GM_RESID
    __bf16* __restrict__ bf_out,      // GM_RELU
    float* __restrict__ Qb,           // GM_QKV
    float* __restrict__ Kc,
    float* __restrict__ Vc,
    int S, int smax) {
  __shared__ __bf16 As[TM][A_STRIDE];        // 128*40*2  = 10240 B
  __shared__ __bf16 BsF[128 * B_STRIDE];     // 128*24*2  =  6144 B

  const int tid  = threadIdx.x;
  const int lane = tid & 31;
  const int wave = tid >> 5;                 // 0..3
  const int mBase = blockIdx.y * TM;
  const int nBase = blockIdx.x * TN;

  v8f acc[2][4] = {};

  // A-fragment addressing (ISA: lanes 0-15 K base 0, lanes 16-31 K base 8;
  // per lane two contiguous runs k=[kh,kh+8) and [kh+16,kh+24)).
  const int khalf = (lane >> 4) * 8;
  const int mr0 = wave * 16 + (lane & 15);
  const int mr1 = (wave + 4) * 16 + (lane & 15);

  // B staging: thread <-> fragment slot (j = tid>>5, lane' = tid&31).
  const int bj    = tid >> 5;
  const int bln   = tid & 31;
  const int bcol  = nBase + bj * 16 + (bln & 15);
  const int brow0 = (bln >> 4) * 8;

  for (int k0 = 0; k0 < K; k0 += TK) {
    // ---- stage A (128 x 32 halfs): 4 x 16B per thread, rows clamped ------
    #pragma unroll
    for (int i = 0; i < 4; ++i) {
      int idx = tid + i * 128;               // 0..511
      int r = idx >> 2;
      int c = (idx & 3) * 8;
      int gm = mBase + r;
      gm = gm < M ? gm : (M - 1);            // clamp: masked rows discarded later
      __bf16* ldst = &As[r][c];
#if HAS_ASYNC_LDS
      __builtin_amdgcn_global_load_async_to_lds_b128(
          (gas_v4i)(void*)(A + (size_t)gm * lda + k0 + c),
          (las_v4i)(void*)ldst, 0, 0);
#else
      *(uint4*)ldst = *(const uint4*)(A + (size_t)gm * lda + k0 + c);
#endif
    }
    // ---- stage B transposed into fragment order: 16 loads w/ immediate
    //      offsets from one base pointer (NN is compile-time) --------------
    {
      const __bf16* bbase = Bw + (size_t)(k0 + brow0) * NN + bcol;
      union { __bf16 h[16]; uint4 x[2]; } tb;
      #pragma unroll
      for (int e = 0; e < 16; ++e)
        tb.h[e] = bbase[(size_t)(e + (e & 8)) * NN];
      __bf16* dst = &BsF[tid * B_STRIDE];
      *(uint4*)dst       = tb.x[0];
      *(uint4*)(dst + 8) = tb.x[1];
    }
    // ---- prefetch next K-step of B (global_prefetch_b8) -----------------
    if (k0 + TK < K)
      __builtin_prefetch(Bw + (size_t)(k0 + TK + (tid >> 2)) * NN + nBase + ((tid & 3) << 4), 0, 1);
#if HAS_ASYNC_LDS
    asm volatile("s_wait_asynccnt 0" ::: "memory");
#endif
    __syncthreads();

    // ---- fragments + WMMA ----------------------------------------------
    union { v16bf v; uint4 x[2]; } fa0, fa1;
    fa0.x[0] = *(const uint4*)&As[mr0][khalf];
    fa0.x[1] = *(const uint4*)&As[mr0][khalf + 16];
    fa1.x[0] = *(const uint4*)&As[mr1][khalf];
    fa1.x[1] = *(const uint4*)&As[mr1][khalf + 16];
    #pragma unroll
    for (int j = 0; j < 4; ++j) {
      union { v16bf v; uint4 x[2]; } fb;
      const __bf16* bp = &BsF[(j * 32 + lane) * B_STRIDE];
      fb.x[0] = *(const uint4*)bp;
      fb.x[1] = *(const uint4*)(bp + 8);
      acc[0][j] = __builtin_amdgcn_wmma_f32_16x16x32_bf16(
          false, fa0.v, false, fb.v, (short)0, acc[0][j], false, false);
      acc[1][j] = __builtin_amdgcn_wmma_f32_16x16x32_bf16(
          false, fa1.v, false, fb.v, (short)0, acc[1][j], false, false);
    }
    __syncthreads();
  }

  // ---- epilogue (C/D: lane 0-15 N=lane M=r; lane 16-31 N=lane-16 M=8+r) --
  #pragma unroll
  for (int s2 = 0; s2 < 2; ++s2) {
    const int mTop = mBase + (wave + s2 * 4) * 16 + (lane >> 4) * 8;
    #pragma unroll
    for (int j = 0; j < 4; ++j) {
      const int n = nBase + j * 16 + (lane & 15);
      const float bv = bias[n];
      #pragma unroll
      for (int r = 0; r < 8; ++r) {
        const int m = mTop + r;
        if (m >= M) continue;
        const float val = acc[s2][j][r] + bv;
        if (MODE == GM_QKV) {
          const int bb = m / S;
          const int s  = m - bb * S;
          if (n < DMODEL) {
            Qb[(((size_t)bb * NHEAD + (n / HDIM)) * smax + s) * HDIM + (n % HDIM)] = val;
          } else if (n < 2 * DMODEL) {
            const int nn = n - DMODEL;
            Kc[(((size_t)bb * NHEAD + (nn / HDIM)) * smax + s) * HDIM + (nn % HDIM)] = val;
          } else {
            const int nn = n - 2 * DMODEL;
            Vc[(((size_t)bb * NHEAD + (nn / HDIM)) * smax + s) * HDIM + (nn % HDIM)] = val;
          }
        } else if (MODE == GM_RESID) {
          resid_out[(size_t)m * NN + n] += val;
        } else { // GM_RELU -> bf16
          bf_out[(size_t)m * NN + n] = (__bf16)fmaxf(val, 0.f);
        }
      }
    }
  }
}

// ---------------------------------------------------------------------------
// Prefill causal attention, online softmax. One thread per (b,h,q).
// ---------------------------------------------------------------------------
__global__ __launch_bounds__(128) void k_attn_prefill(
    const float* __restrict__ Qb, const float* __restrict__ Kc,
    const float* __restrict__ Vc, __bf16* __restrict__ ctx_bf,
    int S, int smax) {
  const int gid = blockIdx.x * blockDim.x + threadIdx.x;
  if (gid >= BATCH * NHEAD * S) return;
  const int q  = gid % S;
  const int bh = gid / S;

  const float* qv = Qb + ((size_t)bh * smax + q) * HDIM;
  float qr[HDIM];
  #pragma unroll
  for (int d = 0; d < HDIM; ++d) qr[d] = qv[d] * ATTN_SCALE;

  float mmax = -3.4e38f, lsum = 0.f;
  float accv[HDIM];
  #pragma unroll
  for (int d = 0; d < HDIM; ++d) accv[d] = 0.f;

  const float* Kp = Kc + (size_t)bh * smax * HDIM;
  const float* Vp = Vc + (size_t)bh * smax * HDIM;
  for (int s = 0; s <= q; ++s) {
    float sc = 0.f;
    #pragma unroll
    for (int d = 0; d < HDIM; ++d) sc = fmaf(qr[d], Kp[(size_t)s * HDIM + d], sc);
    const float nm   = fmaxf(mmax, sc);
    const float corr = __expf(mmax - nm);
    const float w    = __expf(sc - nm);
    lsum = lsum * corr + w;
    #pragma unroll
    for (int d = 0; d < HDIM; ++d)
      accv[d] = fmaf(w, Vp[(size_t)s * HDIM + d], accv[d] * corr);
    mmax = nm;
  }
  const float inv = 1.f / lsum;
  const int b = bh / NHEAD, h = bh % NHEAD;
  __bf16* out = ctx_bf + ((size_t)(b * S + q)) * DMODEL + h * HDIM;
  #pragma unroll
  for (int d = 0; d < HDIM; ++d) out[d] = (__bf16)(accv[d] * inv);
}

// ---------------------------------------------------------------------------
// Decode GEMV: out[b,n] = f(sum_k in[b,k]*W[k,n] + bias[n]). fp32 path.
// ---------------------------------------------------------------------------
__global__ __launch_bounds__(256) void k_dec_mm(
    const float* __restrict__ in, const float* __restrict__ W,
    const float* __restrict__ bias, int K, int N, int mode,
    float* __restrict__ out,
    float* __restrict__ Kc, float* __restrict__ Vc,
    int s_new, int smax) {
  const int n = blockIdx.x * 256 + threadIdx.x;
  const int b = blockIdx.y;
  if (n >= N) return;
  const float* ip = in + (size_t)b * K;
  float acc = bias[n];
  #pragma unroll 4
  for (int k = 0; k < K; ++k) acc = fmaf(ip[k], W[(size_t)k * N + n], acc);

  if (mode == GM_QKV) {
    if (n < DMODEL) {
      out[(size_t)b * DMODEL + n] = acc;   // q
    } else if (n < 2 * DMODEL) {
      const int nn = n - DMODEL;
      Kc[(((size_t)b * NHEAD + nn / HDIM) * smax + s_new) * HDIM + nn % HDIM] = acc;
    } else {
      const int nn = n - 2 * DMODEL;
      Vc[(((size_t)b * NHEAD + nn / HDIM) * smax + s_new) * HDIM + nn % HDIM] = acc;
    }
  } else if (mode == GM_RELU) {
    out[(size_t)b * N + n] = fmaxf(acc, 0.f);
  } else { // GM_RESID
    out[(size_t)b * N + n] += acc;
  }
}

// ---------------------------------------------------------------------------
// Decode attention: 32 threads, one per (b,h); attends over cached K/V.
// ---------------------------------------------------------------------------
__global__ void k_dec_attn(const float* __restrict__ qdec,
                           const float* __restrict__ Kc,
                           const float* __restrict__ Vc,
                           float* __restrict__ ctx, int slen, int smax) {
  const int t = threadIdx.x;
  if (t >= BATCH * NHEAD) return;
  const int b = t >> 4, h = t & 15;
  const float* q = qdec + (size_t)b * DMODEL + h * HDIM;
  float qr[HDIM];
  #pragma unroll
  for (int d = 0; d < HDIM; ++d) qr[d] = q[d] * ATTN_SCALE;

  float mmax = -3.4e38f, lsum = 0.f;
  float accv[HDIM];
  #pragma unroll
  for (int d = 0; d < HDIM; ++d) accv[d] = 0.f;

  const float* Kp = Kc + (size_t)(b * NHEAD + h) * smax * HDIM;
  const float* Vp = Vc + (size_t)(b * NHEAD + h) * smax * HDIM;
  for (int s = 0; s < slen; ++s) {
    float sc = 0.f;
    #pragma unroll
    for (int d = 0; d < HDIM; ++d) sc = fmaf(qr[d], Kp[(size_t)s * HDIM + d], sc);
    const float nm   = fmaxf(mmax, sc);
    const float corr = __expf(mmax - nm);
    const float w    = __expf(sc - nm);
    lsum = lsum * corr + w;
    #pragma unroll
    for (int d = 0; d < HDIM; ++d)
      accv[d] = fmaf(w, Vp[(size_t)s * HDIM + d], accv[d] * corr);
    mmax = nm;
  }
  const float inv = 1.f / lsum;
  #pragma unroll
  for (int d = 0; d < HDIM; ++d)
    ctx[(size_t)b * DMODEL + h * HDIM + d] = accv[d] * inv;
}

// ---------------------------------------------------------------------------
// Decoder head: p = gelu(h@dec1+b1)@dec2+b2 -> preds; nxt token -> xdec.
// ---------------------------------------------------------------------------
__global__ __launch_bounds__(256) void k_head(
    const float* __restrict__ xin, long xstride,
    const float* __restrict__ dec1W, const float* __restrict__ dec1b,
    const float* __restrict__ dec2W, const float* __restrict__ dec2b,
    const float* __restrict__ posW, const float* __restrict__ posb,
    const float* __restrict__ timev,
    float* __restrict__ out_pred, int t, float* __restrict__ xdec) {
  const int b = blockIdx.x;
  const int tid = threadIdx.x;
  __shared__ float hs[DMODEL];
  __shared__ float gs[DMODEL];
  __shared__ float p[2];

  const float* xr = xin + (size_t)b * xstride;
  for (int d = tid; d < DMODEL; d += 256) hs[d] = xr[d];
  __syncthreads();

  for (int n = tid; n < DMODEL; n += 256) {
    float a = dec1b[n];
    for (int k = 0; k < DMODEL; ++k) a = fmaf(hs[k], dec1W[(size_t)k * DMODEL + n], a);
    gs[n] = 0.5f * a * (1.f + erff(a * 0.7071067811865475f));  // exact gelu
  }
  __syncthreads();

  if (tid < 2) {
    float a = dec2b[tid];
    for (int k = 0; k < DMODEL; ++k) a = fmaf(gs[k], dec2W[(size_t)k * 2 + tid], a);
    p[tid] = a;
    out_pred[((size_t)b * T_DEC + t) * 2 + tid] = a;
  }
  __syncthreads();

  for (int n = tid; n < DMODEL; n += 256)
    xdec[(size_t)b * DMODEL + n] =
        p[0] * posW[n] + p[1] * posW[DMODEL + n] + posb[n] + timev[n];
}

// ---------------------------------------------------------------------------
// Host: orchestrate prefill (WMMA GEMMs) + KV-cached decode.
// ---------------------------------------------------------------------------
extern "C" void kernel_launch(void* const* d_in, const int* in_sizes, int n_in,
                              void* d_out, int out_size, void* d_ws, size_t ws_size,
                              hipStream_t stream) {
  (void)in_sizes; (void)n_in; (void)out_size; (void)ws_size;

  const float* feats    = (const float*)d_in[0];
  const float* past     = (const float*)d_in[1];
  const int*   intent   = (const int*)d_in[2];
  const float* img_pos  = (const float*)d_in[3];
  const float* time_e   = (const float*)d_in[4];
  const float* intent_e = (const float*)d_in[5];
  const float* past_W   = (const float*)d_in[6];
  const float* past_b   = (const float*)d_in[7];
  const float* pos_W    = (const float*)d_in[8];
  const float* pos_b    = (const float*)d_in[9];
  const float* ln1_g    = (const float*)d_in[10];
  const float* ln1_b    = (const float*)d_in[11];
  const float* qkv_W    = (const float*)d_in[12];
  const float* qkv_b    = (const float*)d_in[13];
  const float* out_W    = (const float*)d_in[14];
  const float* out_b    = (const float*)d_in[15];
  const float* ln2_g    = (const float*)d_in[16];
  const float* ln2_b    = (const float*)d_in[17];
  const float* ff1_W    = (const float*)d_in[18];
  const float* ff1_b    = (const float*)d_in[19];
  const float* ff2_W    = (const float*)d_in[20];
  const float* ff2_b    = (const float*)d_in[21];
  const float* dec1_W   = (const float*)d_in[22];
  const float* dec1_b   = (const float*)d_in[23];
  const float* dec2_W   = (const float*)d_in[24];
  const float* dec2_b   = (const float*)d_in[25];
  float* outp = (float*)d_out;

  // ---- workspace carve ----
  const size_t SZ_WQKV = (size_t)NLAYER * DMODEL * 3 * DMODEL;   // halfs
  const size_t SZ_WOUT = (size_t)NLAYER * DMODEL * DMODEL;
  const size_t SZ_WFF1 = (size_t)NLAYER * DMODEL * 4 * DMODEL;
  const size_t SZ_WFF2 = (size_t)NLAYER * 4 * DMODEL * DMODEL;
  const size_t CACHE_L = (size_t)BATCH * NHEAD * SMAX * HDIM;    // floats/layer

  char* p = (char*)d_ws;
  auto carve = [&](size_t bytes) -> void* {
    void* r = (void*)p;
    p += (bytes + 255) & ~(size_t)255;
    return r;
  };
  __bf16* wqkv = (__bf16*)carve(SZ_WQKV * 2);
  __bf16* wout = (__bf16*)carve(SZ_WOUT * 2);
  __bf16* wff1 = (__bf16*)carve(SZ_WFF1 * 2);
  __bf16* wff2 = (__bf16*)carve(SZ_WFF2 * 2);
  float*  xf32 = (float*)carve((size_t)M0 * DMODEL * 4);
  __bf16* hbf  = (__bf16*)carve((size_t)M0 * DMODEL * 2);
  __bf16* hbf2 = (__bf16*)carve((size_t)M0 * 4 * DMODEL * 2);
  float*  Qb   = (float*)carve((size_t)BATCH * NHEAD * SMAX * HDIM * 4);
  float*  Kc   = (float*)carve((size_t)NLAYER * CACHE_L * 4);
  float*  Vc   = (float*)carve((size_t)NLAYER * CACHE_L * 4);
  float*  xdec = (float*)carve((size_t)BATCH * DMODEL * 4);
  float*  hdec = (float*)carve((size_t)BATCH * DMODEL * 4);
  float*  qdec = (float*)carve((size_t)BATCH * DMODEL * 4);
  float*  ctxd = (float*)carve((size_t)BATCH * DMODEL * 4);
  float*  fdec = (float*)carve((size_t)BATCH * 4 * DMODEL * 4);

  // ---- weights -> bf16 (once per launch; deterministic) ----
  k_f2bf<<<1024, 256, 0, stream>>>(qkv_W, wqkv, (long)SZ_WQKV);
  k_f2bf<<<1024, 256, 0, stream>>>(out_W, wout, (long)SZ_WOUT);
  k_f2bf<<<1024, 256, 0, stream>>>(ff1_W, wff1, (long)SZ_WFF1);
  k_f2bf<<<1024, 256, 0, stream>>>(ff2_W, wff2, (long)SZ_WFF2);

  // ---- embed the initial sequence ----
  k_embed<<<dim3(S0, BATCH), 256, 0, stream>>>(feats, past, intent, img_pos,
                                               time_e, intent_e, past_W, past_b, xf32);

  const int gM = (M0 + TM - 1) / TM;  // 17

  // ---- prefill: 3 encoder layers with WMMA GEMMs ----
  for (int l = 0; l < NLAYER; ++l) {
    float* KcL = Kc + (size_t)l * CACHE_L;
    float* VcL = Vc + (size_t)l * CACHE_L;

    k_ln<__bf16><<<M0, 256, 0, stream>>>(xf32, ln1_g + l * DMODEL, ln1_b + l * DMODEL, hbf);
    k_gemm<3 * DMODEL, GM_QKV><<<dim3(3 * DMODEL / TN, gM), 128, 0, stream>>>(
        hbf, DMODEL, wqkv + (size_t)l * DMODEL * 3 * DMODEL,
        qkv_b + (size_t)l * 3 * DMODEL, M0, DMODEL,
        nullptr, nullptr, Qb, KcL, VcL, S0, SMAX);
    k_attn_prefill<<<(BATCH * NHEAD * S0 + 127) / 128, 128, 0, stream>>>(
        Qb, KcL, VcL, hbf, S0, SMAX);
    k_gemm<DMODEL, GM_RESID><<<dim3(DMODEL / TN, gM), 128, 0, stream>>>(
        hbf, DMODEL, wout + (size_t)l * DMODEL * DMODEL,
        out_b + (size_t)l * DMODEL, M0, DMODEL,
        xf32, nullptr, nullptr, nullptr, nullptr, S0, SMAX);
    k_ln<__bf16><<<M0, 256, 0, stream>>>(xf32, ln2_g + l * DMODEL, ln2_b + l * DMODEL, hbf);
    k_gemm<4 * DMODEL, GM_RELU><<<dim3(4 * DMODEL / TN, gM), 128, 0, stream>>>(
        hbf, DMODEL, wff1 + (size_t)l * DMODEL * 4 * DMODEL,
        ff1_b + (size_t)l * 4 * DMODEL, M0, DMODEL,
        nullptr, hbf2, nullptr, nullptr, nullptr, S0, SMAX);
    k_gemm<DMODEL, GM_RESID><<<dim3(DMODEL / TN, gM), 128, 0, stream>>>(
        hbf2, 4 * DMODEL, wff2 + (size_t)l * 4 * DMODEL * DMODEL,
        ff2_b + (size_t)l * DMODEL, M0, 4 * DMODEL,
        xf32, nullptr, nullptr, nullptr, nullptr, S0, SMAX);
  }

  // ---- head for t=0 (last prefill token) ----
  k_head<<<BATCH, 256, 0, stream>>>(xf32 + (size_t)(S0 - 1) * DMODEL, (long)S0 * DMODEL,
                                    dec1_W, dec1_b, dec2_W, dec2_b, pos_W, pos_b,
                                    time_e + (size_t)N_PAST * DMODEL, outp, 0, xdec);

  // ---- decode steps 1..19 (KV-cached single-token passes) ----
  for (int t = 1; t < T_DEC; ++t) {
    const int s_new = S0 + t - 1;
    const int slen  = s_new + 1;
    for (int l = 0; l < NLAYER; ++l) {
      float* KcL = Kc + (size_t)l * CACHE_L;
      float* VcL = Vc + (size_t)l * CACHE_L;

      k_ln<float><<<BATCH, 256, 0, stream>>>(xdec, ln1_g + l * DMODEL, ln1_b + l * DMODEL, hdec);
      k_dec_mm<<<dim3(9, BATCH), 256, 0, stream>>>(
          hdec, qkv_W + (size_t)l * DMODEL * 3 * DMODEL, qkv_b + (size_t)l * 3 * DMODEL,
          DMODEL, 3 * DMODEL, GM_QKV, qdec, KcL, VcL, s_new, SMAX);
      k_dec_attn<<<1, 32, 0, stream>>>(qdec, KcL, VcL, ctxd, slen, SMAX);
      k_dec_mm<<<dim3(3, BATCH), 256, 0, stream>>>(
          ctxd, out_W + (size_t)l * DMODEL * DMODEL, out_b + (size_t)l * DMODEL,
          DMODEL, DMODEL, GM_RESID, xdec, nullptr, nullptr, 0, SMAX);
      k_ln<float><<<BATCH, 256, 0, stream>>>(xdec, ln2_g + l * DMODEL, ln2_b + l * DMODEL, hdec);
      k_dec_mm<<<dim3(12, BATCH), 256, 0, stream>>>(
          hdec, ff1_W + (size_t)l * DMODEL * 4 * DMODEL, ff1_b + (size_t)l * 4 * DMODEL,
          DMODEL, 4 * DMODEL, GM_RELU, fdec, nullptr, nullptr, 0, SMAX);
      k_dec_mm<<<dim3(3, BATCH), 256, 0, stream>>>(
          fdec, ff2_W + (size_t)l * 4 * DMODEL * DMODEL, ff2_b + (size_t)l * DMODEL,
          4 * DMODEL, DMODEL, GM_RESID, xdec, nullptr, nullptr, 0, SMAX);
    }
    k_head<<<BATCH, 256, 0, stream>>>(xdec, DMODEL, dec1_W, dec1_b, dec2_W, dec2_b,
                                      pos_W, pos_b, time_e + (size_t)(N_PAST + t) * DMODEL,
                                      outp, t, xdec);
  }
}
